// MambaResidualBlock_67980742362029
// MI455X (gfx1250) — matrix operand order repaired and checked by reading, not compiled, not run on hardware
//
#include <hip/hip_runtime.h>
#include <hip/hip_bf16.h>
#include <math.h>

// ---------------------------------------------------------------------------
// Problem constants
// ---------------------------------------------------------------------------
#define BB      4
#define LL      1024
#define DM      1024
#define DI      2048          // EXPAND * DM
#define DSTATE  16
#define DTR     64            // ceil(DM/16)
#define DCONV   4
#define MT      (BB*LL)       // 4096 tokens

typedef __attribute__((ext_vector_type(16))) __bf16 v16bf;
typedef __attribute__((ext_vector_type(8)))  float  v8f;

union AFrag { v16bf v; uint4 q[2]; };

// ---------------------------------------------------------------------------
// Generic f32 -> bf16 cast (grid-stride)
// ---------------------------------------------------------------------------
__global__ void mamba_cast_bf16(const float* __restrict__ in,
                                __bf16* __restrict__ out, size_t n) {
    size_t i = (size_t)blockIdx.x * blockDim.x + threadIdx.x;
    size_t stride = (size_t)gridDim.x * blockDim.x;
    for (; i < n; i += stride) out[i] = (__bf16)in[i];
}

// Strided cast: dt_r = dbl[:, 0:64] (ld 96) -> bf16 [MT x 64]
__global__ void mamba_cast_dtr(const float* __restrict__ dbl,
                               __bf16* __restrict__ out) {
    int i = blockIdx.x * blockDim.x + threadIdx.x;
    if (i >= MT * DTR) return;
    int r = i / DTR, c = i % DTR;
    out[i] = (__bf16)dbl[(size_t)r * 96 + c];
}

// ---------------------------------------------------------------------------
// RMSNorm over D=1024, write bf16
// ---------------------------------------------------------------------------
__global__ __launch_bounds__(256) void mamba_rmsnorm(
    const float* __restrict__ x, const float* __restrict__ w,
    __bf16* __restrict__ out) {
    __shared__ float red[256];
    int row = blockIdx.x, tid = threadIdx.x;
    const float* xr = x + (size_t)row * DM;
    float s = 0.f;
    #pragma unroll
    for (int i = 0; i < DM / 256; i++) { float v = xr[tid + i * 256]; s += v * v; }
    red[tid] = s; __syncthreads();
    for (int o = 128; o > 0; o >>= 1) {
        if (tid < o) red[tid] += red[tid + o];
        __syncthreads();
    }
    float scale = rsqrtf(red[0] / (float)DM + 1e-5f);
    #pragma unroll
    for (int i = 0; i < DM / 256; i++) {
        int c = tid + i * 256;
        out[(size_t)row * DM + c] = (__bf16)(xr[c] * scale * w[c]);
    }
}

// ---------------------------------------------------------------------------
// WMMA TN GEMM (small-tile): C[M,N] = A[M,K] * W[N,K]^T
// wave tile 32x32 (2x2 WMMA); block tile 64x128; used for N=96 GEMM
// ---------------------------------------------------------------------------
__global__ __launch_bounds__(256) void mamba_gemm_bf16(
    const __bf16* __restrict__ A, const __bf16* __restrict__ W,
    float* __restrict__ C, int M, int N, int K) {
    int tid  = threadIdx.x;
    int lane = tid & 31;
    int wid  = tid >> 5;
    int wm   = wid >> 2;            // 0..1
    int wn   = wid & 3;             // 0..3
    int m0 = blockIdx.y * 64  + wm * 32;
    int n0 = blockIdx.x * 128 + wn * 32;
    if (m0 >= M || n0 >= N) return;
    int hi = lane >> 4;             // 0/1
    int lo = lane & 15;

    v8f acc[2][2];
    #pragma unroll
    for (int i = 0; i < 2; i++)
        #pragma unroll
        for (int j = 0; j < 2; j++)
            #pragma unroll
            for (int r = 0; r < 8; r++) acc[i][j][r] = 0.f;

    const __bf16* arow[2] = { A + (size_t)(m0 + lo) * K,
                              A + (size_t)(m0 + 16 + lo) * K };
    const __bf16* brow[2] = { W + (size_t)(n0 + lo) * K,
                              W + (size_t)(n0 + 16 + lo) * K };

    for (int k0 = 0; k0 < K; k0 += 32) {
        AFrag a[2], b[2];
        #pragma unroll
        for (int i = 0; i < 2; i++) {
            // A layout: lane(hi=0): K 0..7 & 16..23 ; hi=1: K 8..15 & 24..31
            a[i].q[0] = *(const uint4*)(arow[i] + k0 + hi * 8);
            a[i].q[1] = *(const uint4*)(arow[i] + k0 + 16 + hi * 8);
            // B layout: same as A (hi=0: K 0..7 & 16..23 ; hi=1: K 8..15 & 24..31)
            b[i].q[0] = *(const uint4*)(brow[i] + k0 + hi * 8);
            b[i].q[1] = *(const uint4*)(brow[i] + k0 + 16 + hi * 8);
        }
        #pragma unroll
        for (int i = 0; i < 2; i++)
            #pragma unroll
            for (int j = 0; j < 2; j++)
                acc[i][j] = __builtin_amdgcn_wmma_f32_16x16x32_bf16(
                    false, a[i].v, false, b[j].v, (short)0, acc[i][j],
                    false, false);
    }

    #pragma unroll
    for (int i = 0; i < 2; i++)
        #pragma unroll
        for (int j = 0; j < 2; j++) {
            float* cp = C + (size_t)(m0 + i * 16 + hi * 8) * N + (n0 + j * 16 + lo);
            #pragma unroll
            for (int r = 0; r < 8; r++) cp[(size_t)r * N] = acc[i][j][r];
        }
}

// ---------------------------------------------------------------------------
// WMMA TN GEMM (big-tile): wave tile 64x64 (4x4 WMMA), block tile 128x256.
// 16 WMMAs per 16 b128 loads per K-step -> 2x arithmetic intensity from L2.
// Requires M % 128 == 0, N % 256 == 0, K % 32 == 0.
// ---------------------------------------------------------------------------
__global__ __launch_bounds__(256, 1) void mamba_gemm_bf16_big(
    const __bf16* __restrict__ A, const __bf16* __restrict__ W,
    float* __restrict__ C, int M, int N, int K) {
    int tid  = threadIdx.x;
    int lane = tid & 31;
    int wid  = tid >> 5;
    int wm   = wid >> 2;            // 0..1
    int wn   = wid & 3;             // 0..3
    int m0 = blockIdx.y * 128 + wm * 64;
    int n0 = blockIdx.x * 256 + wn * 64;
    int hi = lane >> 4;             // 0/1
    int lo = lane & 15;

    v8f acc[4][4];
    #pragma unroll
    for (int i = 0; i < 4; i++)
        #pragma unroll
        for (int j = 0; j < 4; j++)
            #pragma unroll
            for (int r = 0; r < 8; r++) acc[i][j][r] = 0.f;

    const __bf16* abase = A + (size_t)(m0 + lo) * K;
    const __bf16* bbase = W + (size_t)(n0 + lo) * K;
    const size_t rowK16 = (size_t)16 * K;

    for (int k0 = 0; k0 < K; k0 += 32) {
        AFrag a[4], b[4];
        #pragma unroll
        for (int i = 0; i < 4; i++) {
            const __bf16* ap = abase + (size_t)i * rowK16 + k0;
            a[i].q[0] = *(const uint4*)(ap + hi * 8);
            a[i].q[1] = *(const uint4*)(ap + 16 + hi * 8);
            const __bf16* bp = bbase + (size_t)i * rowK16 + k0;
            b[i].q[0] = *(const uint4*)(bp + hi * 8);
            b[i].q[1] = *(const uint4*)(bp + 16 + hi * 8);
        }
        #pragma unroll
        for (int i = 0; i < 4; i++)
            #pragma unroll
            for (int j = 0; j < 4; j++)
                acc[i][j] = __builtin_amdgcn_wmma_f32_16x16x32_bf16(
                    false, a[i].v, false, b[j].v, (short)0, acc[i][j],
                    false, false);
    }

    #pragma unroll
    for (int i = 0; i < 4; i++)
        #pragma unroll
        for (int j = 0; j < 4; j++) {
            float* cp = C + (size_t)(m0 + i * 16 + hi * 8) * N + (n0 + j * 16 + lo);
            #pragma unroll
            for (int r = 0; r < 8; r++) cp[(size_t)r * N] = acc[i][j][r];
        }
}

// ---------------------------------------------------------------------------
// Depthwise causal conv (k=4) + SiLU; reads u = inproj[:, 0:DI]
// ---------------------------------------------------------------------------
__global__ void mamba_conv_silu(const float* __restrict__ inproj,
                                const float* __restrict__ cw,
                                const float* __restrict__ cb,
                                float* __restrict__ ucf,
                                __bf16* __restrict__ ucb) {
    int idx = blockIdx.x * blockDim.x + threadIdx.x;
    if (idx >= MT * DI) return;
    int d  = idx % DI;
    int bl = idx / DI;
    int l  = bl % LL;
    int b  = bl / LL;
    float acc = cb[d];
    #pragma unroll
    for (int k = 0; k < DCONV; k++) {
        int ls = l - (DCONV - 1) + k;
        if (ls >= 0)
            acc += inproj[((size_t)(b * LL + ls)) * (2 * DI) + d] * cw[d * DCONV + k];
    }
    float v = acc / (1.f + __expf(-acc));   // SiLU
    ucf[idx] = v;
    ucb[idx] = (__bf16)v;
}

// softplus(delta_raw + dt_b) in-place
__global__ void mamba_softplus(float* __restrict__ delta,
                               const float* __restrict__ dtb) {
    int idx = blockIdx.x * blockDim.x + threadIdx.x;
    if (idx >= MT * DI) return;
    float v = delta[idx] + dtb[idx % DI];
    delta[idx] = (v > 20.f) ? v : log1pf(__expf(v));
}

// ---------------------------------------------------------------------------
// Selective scan: one lane per (b,d); 16-state in registers; B_t/C_t via LDS.
// Fused epilogue: y = (scan + u*Dp) * silu(res) -> bf16 for out_proj.
// ---------------------------------------------------------------------------
__global__ __launch_bounds__(256) void mamba_scan(
    const float* __restrict__ ucf, const float* __restrict__ delta,
    const float* __restrict__ dbl, const float* __restrict__ inproj,
    const float* __restrict__ A_log, const float* __restrict__ Dp,
    __bf16* __restrict__ yb, float* __restrict__ hT) {
    __shared__ float BC[32];                 // [0:16)=B_t, [16:32)=C_t
    int b = blockIdx.x >> 3;                 // 4 batches x 8 chunks
    int d = (blockIdx.x & 7) * 256 + threadIdx.x;
    float Arow[DSTATE], h[DSTATE];
    #pragma unroll
    for (int n = 0; n < DSTATE; n++) {
        Arow[n] = -__expf(A_log[(size_t)d * DSTATE + n]);
        h[n] = 0.f;
    }
    float dpd = Dp[d];
    for (int l = 0; l < LL; l++) {
        size_t row = (size_t)b * LL + l;
        __syncthreads();
        if (threadIdx.x < 32) BC[threadIdx.x] = dbl[row * 96 + DTR + threadIdx.x];
        __syncthreads();
        float dlt = delta[row * DI + d];
        float ut  = ucf[row * DI + d];
        float du  = dlt * ut;
        float y = 0.f;
        #pragma unroll
        for (int n = 0; n < DSTATE; n++) {
            h[n] = __expf(dlt * Arow[n]) * h[n] + du * BC[n];
            y += h[n] * BC[DSTATE + n];
        }
        float res = inproj[row * (2 * DI) + DI + d];
        float gate = res / (1.f + __expf(-res));
        yb[row * DI + d] = (__bf16)((y + ut * dpd) * gate);
    }
    #pragma unroll
    for (int n = 0; n < DSTATE; n++)
        hT[((size_t)b * DI + d) * DSTATE + n] = h[n];
}

// out = gemm_out + x (in-place on gemm_out) ; also bf16 copy
__global__ void mamba_resadd(float* __restrict__ o, const float* __restrict__ x,
                             __bf16* __restrict__ ob) {
    int i = blockIdx.x * blockDim.x + threadIdx.x;
    if (i >= MT * DM) return;
    float v = o[i] + x[i];
    o[i] = v;
    ob[i] = (__bf16)v;
}

// h1 = gelu(h1raw + b1) exact -> bf16
__global__ void mamba_gelu(const float* __restrict__ h1raw,
                           const float* __restrict__ b1,
                           __bf16* __restrict__ h1b) {
    int i = blockIdx.x * blockDim.x + threadIdx.x;
    if (i >= MT * DM) return;
    float v = h1raw[i] + b1[i % DM];
    h1b[i] = (__bf16)(0.5f * v * (1.f + erff(v * 0.70710678118654752f)));
}

// z = zraw + b2 + out ; LayerNorm over D=1024 -> d_out
__global__ __launch_bounds__(256) void mamba_ffln(
    const float* __restrict__ zraw, const float* __restrict__ outres,
    const float* __restrict__ b2, const float* __restrict__ lnw,
    const float* __restrict__ lnb, float* __restrict__ dout) {
    __shared__ float rs[256], rs2[256];
    int row = blockIdx.x, tid = threadIdx.x;
    float zv[DM / 256];
    float s = 0.f, s2 = 0.f;
    #pragma unroll
    for (int i = 0; i < DM / 256; i++) {
        int c = tid + i * 256;
        float z = zraw[(size_t)row * DM + c] + b2[c] + outres[(size_t)row * DM + c];
        zv[i] = z; s += z; s2 += z * z;
    }
    rs[tid] = s; rs2[tid] = s2; __syncthreads();
    for (int o = 128; o > 0; o >>= 1) {
        if (tid < o) { rs[tid] += rs[tid + o]; rs2[tid] += rs2[tid + o]; }
        __syncthreads();
    }
    float mu  = rs[0] / (float)DM;
    float var = rs2[0] / (float)DM - mu * mu;
    float k = rsqrtf(var + 1e-5f);
    #pragma unroll
    for (int i = 0; i < DM / 256; i++) {
        int c = tid + i * 256;
        dout[(size_t)row * DM + c] = (zv[i] - mu) * k * lnw[c] + lnb[c];
    }
}

// hidden tail: per b: [hT(2048*16) | u rows L-3..L-1 (3*2048)]
__global__ void mamba_hidden(const float* __restrict__ hT,
                             const float* __restrict__ inproj,
                             float* __restrict__ dout_h) {
    const int PER_B = DI * DSTATE + (DCONV - 1) * DI;   // 38912
    int i = blockIdx.x * blockDim.x + threadIdx.x;
    if (i >= BB * PER_B) return;
    int b = i / PER_B, j = i % PER_B;
    float v;
    if (j < DI * DSTATE) {
        v = hT[(size_t)b * DI * DSTATE + j];
    } else {
        int j2 = j - DI * DSTATE;
        int li = j2 / DI, dd = j2 % DI;
        v = inproj[((size_t)(b * LL + (LL - (DCONV - 1) + li))) * (2 * DI) + dd];
    }
    dout_h[i] = v;
}

// ---------------------------------------------------------------------------
// Host launcher
// ---------------------------------------------------------------------------
extern "C" void kernel_launch(void* const* d_in, const int* in_sizes, int n_in,
                              void* d_out, int out_size, void* d_ws, size_t ws_size,
                              hipStream_t stream) {
    (void)in_sizes; (void)n_in; (void)out_size; (void)ws_size;
    const float* x        = (const float*)d_in[0];
    const float* norm_w   = (const float*)d_in[1];
    const float* in_proj  = (const float*)d_in[2];
    const float* conv_w   = (const float*)d_in[3];
    const float* conv_b   = (const float*)d_in[4];
    const float* x_proj   = (const float*)d_in[5];
    const float* dt_w     = (const float*)d_in[6];
    const float* dt_b     = (const float*)d_in[7];
    const float* A_log    = (const float*)d_in[8];
    const float* Dp       = (const float*)d_in[9];
    const float* out_proj = (const float*)d_in[10];
    const float* ff_w1    = (const float*)d_in[11];
    const float* ff_b1    = (const float*)d_in[12];
    const float* ff_w2    = (const float*)d_in[13];
    const float* ff_b2    = (const float*)d_in[14];
    const float* ff_ln_w  = (const float*)d_in[15];
    const float* ff_ln_b  = (const float*)d_in[16];

    char* ws = (char*)d_ws;
    size_t off = 0;
    auto alloc = [&](size_t bytes) -> char* {
        char* p = ws + off;
        off += (bytes + 255) & ~(size_t)255;
        return p;
    };
    __bf16* wb_inproj  = (__bf16*)alloc((size_t)2*DI*DM*2);          // [4096x1024]
    __bf16* wb_xproj   = (__bf16*)alloc((size_t)96*DI*2);            // [96x2048]
    __bf16* wb_dtw     = (__bf16*)alloc((size_t)DI*DTR*2);           // [2048x64]
    __bf16* wb_outproj = (__bf16*)alloc((size_t)DM*DI*2);            // [1024x2048]
    __bf16* wb_ff1     = (__bf16*)alloc((size_t)DM*DM*2);
    __bf16* wb_ff2     = (__bf16*)alloc((size_t)DM*DM*2);
    __bf16* xn_b       = (__bf16*)alloc((size_t)MT*DM*2);
    float*  dbl        = (float*) alloc((size_t)MT*96*4);
    __bf16* dtr_b      = (__bf16*)alloc((size_t)MT*DTR*2);
    float*  hT         = (float*) alloc((size_t)BB*DI*DSTATE*4);
    float*  inproj_o   = (float*) alloc((size_t)MT*2*DI*4);          // 64MB, alive to end
    float*  ucf        = (float*) alloc((size_t)MT*DI*4);            // 32MB -> reuse h1raw
    __bf16* ucb        = (__bf16*)alloc((size_t)MT*DI*2);            // 16MB -> reuse y_b
    float*  delta      = (float*) alloc((size_t)MT*DI*4);            // 32MB -> reuse outraw+zraw
    __bf16* obh1       = (__bf16*)alloc((size_t)MT*DM*2*2);          // out_b | h1_b
    // lifetime reuse
    float*  h1raw  = ucf;
    __bf16* y_b    = ucb;
    float*  outraw = delta;                       // out f32 [4096x1024]
    float*  zraw   = delta + (size_t)MT*DM;       // z  f32 [4096x1024]
    __bf16* out_b  = obh1;
    __bf16* h1_b   = obh1 + (size_t)MT*DM;

    float* dout_main = (float*)d_out;                     // [4,1024,1024]
    float* dout_hid  = dout_main + (size_t)MT*DM;         // [4,38912]

    const int T = 256;
    // --- weight casts to bf16 ---
    mamba_cast_bf16<<<4096, T, 0, stream>>>(in_proj,  wb_inproj,  (size_t)2*DI*DM);
    mamba_cast_bf16<<<768,  T, 0, stream>>>(x_proj,   wb_xproj,   (size_t)96*DI);
    mamba_cast_bf16<<<512,  T, 0, stream>>>(dt_w,     wb_dtw,     (size_t)DI*DTR);
    mamba_cast_bf16<<<2048, T, 0, stream>>>(out_proj, wb_outproj, (size_t)DM*DI);
    mamba_cast_bf16<<<1024, T, 0, stream>>>(ff_w1,    wb_ff1,     (size_t)DM*DM);
    mamba_cast_bf16<<<1024, T, 0, stream>>>(ff_w2,    wb_ff2,     (size_t)DM*DM);

    // --- RMSNorm -> bf16 ---
    mamba_rmsnorm<<<MT, T, 0, stream>>>(x, norm_w, xn_b);

    // --- in_proj GEMM: [4096x1024]x[4096x1024]^T -> [4096x4096] ---
    mamba_gemm_bf16_big<<<dim3((2*DI)/256, MT/128), T, 0, stream>>>(
        xn_b, wb_inproj, inproj_o, MT, 2*DI, DM);

    // --- conv + silu ---
    mamba_conv_silu<<<(MT*DI)/T, T, 0, stream>>>(inproj_o, conv_w, conv_b, ucf, ucb);

    // --- x_proj GEMM: [4096x2048]x[96x2048]^T -> [4096x96] (small tile, N=96) ---
    mamba_gemm_bf16<<<dim3(1, MT/64), T, 0, stream>>>(ucb, wb_xproj, dbl, MT, 96, DI);

    // --- dt_r cast, delta GEMM, softplus ---
    mamba_cast_dtr<<<(MT*DTR)/T, T, 0, stream>>>(dbl, dtr_b);
    mamba_gemm_bf16_big<<<dim3(DI/256, MT/128), T, 0, stream>>>(
        dtr_b, wb_dtw, delta, MT, DI, DTR);
    mamba_softplus<<<(MT*DI)/T, T, 0, stream>>>(delta, dt_b);

    // --- selective scan (fused gating) ---
    mamba_scan<<<BB*8, T, 0, stream>>>(ucf, delta, dbl, inproj_o, A_log, Dp, y_b, hT);

    // --- out_proj GEMM + residual ---
    mamba_gemm_bf16_big<<<dim3(DM/256, MT/128), T, 0, stream>>>(
        y_b, wb_outproj, outraw, MT, DM, DI);
    mamba_resadd<<<(MT*DM)/T, T, 0, stream>>>(outraw, x, out_b);

    // --- FF1 + GELU ---
    mamba_gemm_bf16_big<<<dim3(DM/256, MT/128), T, 0, stream>>>(
        out_b, wb_ff1, h1raw, MT, DM, DM);
    mamba_gelu<<<(MT*DM)/T, T, 0, stream>>>(h1raw, ff_b1, h1_b);

    // --- FF2 + LayerNorm -> d_out ---
    mamba_gemm_bf16_big<<<dim3(DM/256, MT/128), T, 0, stream>>>(
        h1_b, wb_ff2, zraw, MT, DM, DM);
    mamba_ffln<<<MT, T, 0, stream>>>(zraw, outraw, ff_b2, ff_ln_w, ff_ln_b, dout_main);

    // --- hidden tail ---
    mamba_hidden<<<(BB*(DI*DSTATE+(DCONV-1)*DI) + T - 1)/T, T, 0, stream>>>(
        hT, inproj_o, dout_hid);
}
